// TriangleMultiplicationV43_4861902979616
// MI455X (gfx1250) — matrix operand order repaired and checked by reading, not compiled, not run on hardware
//
#include <hip/hip_runtime.h>
#include <hip/hip_bf16.h>
#include <cstdint>

// ---------------------------------------------------------------------------
// Triangle multiplication (AlphaFold) for MI455X / gfx1250.
// B=1, N=384, C=H=128.  All GEMMs via v_wmma_f32_16x16x32_bf16 (wave32).
// Bandwidth-bound op: bf16 intermediates, L2-resident operands for the
// per-channel triangle GEMMs (192MB L2 >> 75MB of bf16 operands).
// ---------------------------------------------------------------------------

#define NN   384
#define CC   128
#define HH   128
#define N2   (NN * NN)      // 147456
#define LDA  136            // padded LDS stride (bf16 elems) -> conflict-free frag loads
#define LDO  132            // padded LDS stride (f32) for output tile

typedef __attribute__((ext_vector_type(16))) __bf16 v16bf;
typedef __attribute__((ext_vector_type(8)))  __bf16 v8bf;
typedef __attribute__((ext_vector_type(8)))  float  v8f;

__device__ __forceinline__ v16bf cat16(v8bf lo, v8bf hi) {
    return __builtin_shufflevector(lo, hi, 0,1,2,3,4,5,6,7,8,9,10,11,12,13,14,15);
}

// A-fragment (16x32 bf16): lane l<16 -> row l, K {0..7,16..23}; lane>=16 -> K {8..15,24..31}
__device__ __forceinline__ v16bf load_a_frag(const __bf16* base, int stride) {
    const int lane = threadIdx.x & 31;
    const int m    = lane & 15;
    const int klo  = (lane >> 4) << 3;          // 0 or 8
    const __bf16* p = base + m * stride + klo;
    v8bf lo = *(const v8bf*)(p);
    v8bf hi = *(const v8bf*)(p + 16);
    return cat16(lo, hi);
}

// B-fragment (32x16 bf16): lane l<16 -> col l&15, K {0..7,16..23}; lane>=16 -> K {8..15,24..31}
__device__ __forceinline__ v16bf load_b_frag(const __bf16* base, int stride) {
    const int lane = threadIdx.x & 31;
    const int n    = lane & 15;
    const int k0   = (lane >> 4) << 3;          // 0 or 8
    const __bf16* p = base + n * stride + k0;
    v8bf lo = *(const v8bf*)(p);
    v8bf hi = *(const v8bf*)(p + 16);
    return cat16(lo, hi);
}

__device__ __forceinline__ v8f wmma_bf16(v16bf a, v16bf b, v8f c) {
    return __builtin_amdgcn_wmma_f32_16x16x32_bf16(
        /*neg_a=*/false, a, /*neg_b=*/false, b,
        /*c_mod=*/(short)0, c, /*reuse_a=*/false, /*reuse_b=*/false);
}

// ---------------------------------------------------------------------------
// Kernel 0: convert the five (Cin,Cout) f32 weights to bf16 W^T[out][in]
// so B-fragments become two contiguous 8-element loads per lane.
// grid = 5*128 blocks x 128 threads
// ---------------------------------------------------------------------------
__global__ __launch_bounds__(128)
void kw_conv(const float* __restrict__ Wl, const float* __restrict__ Wr,
             const float* __restrict__ Wg, const float* __restrict__ We,
             const float* __restrict__ Wo,
             __bf16* __restrict__ WlT, __bf16* __restrict__ WrT,
             __bf16* __restrict__ WgT, __bf16* __restrict__ WeT,
             __bf16* __restrict__ WoT)
{
    const int o    = blockIdx.x & 127;
    const int widx = blockIdx.x >> 7;
    const int c    = threadIdx.x;
    const float* W = (widx == 0) ? Wl : (widx == 1) ? Wr : (widx == 2) ? Wg
                   : (widx == 3) ? We : Wo;
    __bf16* WT     = (widx == 0) ? WlT : (widx == 1) ? WrT : (widx == 2) ? WgT
                   : (widx == 3) ? WeT : WoT;
    WT[o * 128 + c] = (__bf16)W[c * 128 + o];
}

// ---------------------------------------------------------------------------
// Kernel A: fused dual-LayerNorm + 4 projections for a 16-position tile.
//  left  = LN_l(x)@W_l + x@W_e + b_l + b_e  -> leftA [h][i][k]   (bf16)
//  right = LN_r(x)@W_r + b_r                -> rightT[h][j][k]   (bf16, transposed)
//  gate  = sigmoid(x@W_g + b_g)             -> gateH [h][pos]    (bf16)
// grid = N*N/16 = 9216 blocks x 256 threads (8 waves)
// ---------------------------------------------------------------------------
__global__ __launch_bounds__(256)
void ka_proj(const float* __restrict__ pair,
             const float* __restrict__ gl, const float* __restrict__ bl,
             const float* __restrict__ gr, const float* __restrict__ br,
             const float* __restrict__ bias_l, const float* __restrict__ bias_r,
             const float* __restrict__ bias_g, const float* __restrict__ bias_e,
             const __bf16* __restrict__ WlT, const __bf16* __restrict__ WrT,
             const __bf16* __restrict__ WgT, const __bf16* __restrict__ WeT,
             __bf16* __restrict__ leftA, __bf16* __restrict__ rightT,
             __bf16* __restrict__ gateH)
{
    __shared__ __attribute__((aligned(16))) float  xs[16][CC];
    __shared__ float  mu[16], rsg[16];
    __shared__ __attribute__((aligned(16))) __bf16 al[16][LDA];
    __shared__ __attribute__((aligned(16))) __bf16 ar[16][LDA];
    __shared__ __attribute__((aligned(16))) __bf16 ax[16][LDA];

    const int posbase = blockIdx.x * 16;
    const int t    = threadIdx.x;
    const int wave = t >> 5;
    const int lane = t & 31;

    // Phase 1: coalesced load of the 16x128 f32 tile
    {
        const int row = t >> 4, c0 = (t & 15) * 8;
        const float* src = pair + (size_t)(posbase + row) * CC + c0;
        *(float4*)&xs[row][c0]     = *(const float4*)src;
        *(float4*)&xs[row][c0 + 4] = *(const float4*)(src + 4);
    }
    __syncthreads();

    // Phase 2: per-row mean/var (shared by both LayerNorms).  2 rows per wave.
    {
        const int row = 2 * wave + (lane >> 4);
        const int c0  = (lane & 15) * 8;
        float s = 0.f, ss = 0.f;
        #pragma unroll
        for (int j = 0; j < 8; j++) { float v = xs[row][c0 + j]; s += v; ss += v * v; }
        #pragma unroll
        for (int m = 1; m < 16; m <<= 1) {
            s  += __shfl_xor(s,  m, 32);
            ss += __shfl_xor(ss, m, 32);
        }
        const float mean = s * (1.f / CC);
        const float var  = ss * (1.f / CC) - mean * mean;
        if ((lane & 15) == 0) { mu[row] = mean; rsg[row] = rsqrtf(var + 1e-5f); }
    }
    __syncthreads();

    // Phase 3: normalize + convert to bf16 A-tiles (padded stride)
    {
        const int row = t >> 4, c0 = (t & 15) * 8;
        const float mean = mu[row], rs = rsg[row];
        #pragma unroll
        for (int j = 0; j < 8; j++) {
            const int c = c0 + j;
            const float x  = xs[row][c];
            const float xn = (x - mean) * rs;
            al[row][c] = (__bf16)(xn * gl[c] + bl[c]);
            ar[row][c] = (__bf16)(xn * gr[c] + br[c]);
            ax[row][c] = (__bf16)x;
        }
    }
    __syncthreads();

    // Phase 4: WMMA.  Wave w owns h-tile w (16 output channels), K = 128.
    const int hbase = wave * 16;
    v8f accL = {}, accR = {}, accG = {};
    #pragma unroll
    for (int ks = 0; ks < 4; ks++) {
        const int k = ks * 32;
        v16bf fa_x = load_a_frag(&ax[0][0] + k, LDA);
        v16bf fa_l = load_a_frag(&al[0][0] + k, LDA);
        v16bf fa_r = load_a_frag(&ar[0][0] + k, LDA);
        v16bf fb_l = load_b_frag(WlT + hbase * CC + k, CC);
        v16bf fb_e = load_b_frag(WeT + hbase * CC + k, CC);
        v16bf fb_r = load_b_frag(WrT + hbase * CC + k, CC);
        v16bf fb_g = load_b_frag(WgT + hbase * CC + k, CC);
        accL = wmma_bf16(fa_x, fb_e, accL);   // eq branch fused into left acc
        accL = wmma_bf16(fa_l, fb_l, accL);
        accR = wmma_bf16(fa_r, fb_r, accR);
        accG = wmma_bf16(fa_x, fb_g, accG);
    }

    // Store: D(m,n) -> lane gives n (channel), VGPR r gives m (position)
    const int n    = lane & 15;
    const int h    = hbase + n;
    const int msel = (lane >> 4) * 8;
    const int p    = posbase / NN;            // i (row of pair)
    const int q0   = (posbase % NN) + msel;   // j start for this lane's rows

    const float bL = bias_l[h] + bias_e[h];
    const float bG = bias_g[h];
    const float bR = bias_r[h];

    v8bf outL, outG;
    #pragma unroll
    for (int r = 0; r < 8; r++) {
        outL[r] = (__bf16)(accL[r] + bL);
        const float g = accG[r] + bG;
        outG[r] = (__bf16)(1.f / (1.f + __expf(-g)));
    }
    *(v8bf*)(leftA + (size_t)h * N2 + posbase + msel) = outL;  // contiguous 16B
    *(v8bf*)(gateH + (size_t)h * N2 + posbase + msel) = outG;  // contiguous 16B

    __bf16* rp = rightT + (size_t)h * N2 + (size_t)q0 * NN + p; // transpose store
    #pragma unroll
    for (int r = 0; r < 8; r++) rp[r * NN] = (__bf16)(accR[r] + bR);
}

// ---------------------------------------------------------------------------
// Kernel B: triangle einsum.  128 independent 384x384x384 bf16 GEMMs.
// Operands leftA/rightT (75MB bf16 total) are L2-resident on MI455X (192MB L2),
// so fragments stream directly from global memory -- no LDS staging needed.
// Workgroup tile 64(i) x 128(j) per h; wave = 32x32 (2x2 accumulators).
// grid = (N/64)*(N/128)*H = 2304 blocks x 256 threads
// ---------------------------------------------------------------------------
__global__ __launch_bounds__(256)
void kb_tri(const __bf16* __restrict__ leftA, const __bf16* __restrict__ rightT,
            __bf16* __restrict__ midH)
{
    const int b  = blockIdx.x;
    const int h  = b & 127;
    const int bt = b >> 7;                 // 0..17
    const int jb = bt % 3;                 // N/128
    const int ib = bt / 3;                 // N/64
    const int wave = threadIdx.x >> 5;
    const int i0 = ib * 64  + (wave >> 2) * 32;
    const int j0 = jb * 128 + (wave & 3)  * 32;

    const __bf16* A = leftA  + (size_t)h * N2;
    const __bf16* B = rightT + (size_t)h * N2;

    v8f acc[2][2] = {};
    for (int k = 0; k < NN; k += 32) {
        // prefetch next K-slab (speculative, L2 temporal)
        __builtin_prefetch(A + (size_t)i0 * NN + k + 32, 0, 3);
        __builtin_prefetch(B + (size_t)j0 * NN + k + 32, 0, 3);

        v16bf a0 = load_a_frag(A + (size_t)i0 * NN + k, NN);
        v16bf a1 = load_a_frag(A + (size_t)(i0 + 16) * NN + k, NN);
        v16bf b0 = load_b_frag(B + (size_t)j0 * NN + k, NN);
        v16bf b1 = load_b_frag(B + (size_t)(j0 + 16) * NN + k, NN);
        acc[0][0] = wmma_bf16(a0, b0, acc[0][0]);
        acc[0][1] = wmma_bf16(a0, b1, acc[0][1]);
        acc[1][0] = wmma_bf16(a1, b0, acc[1][0]);
        acc[1][1] = wmma_bf16(a1, b1, acc[1][1]);
    }

    // Coalesced store: per r, lanes 0-15 write 16 contiguous bf16 (row m),
    // lanes 16-31 write row m+8.
    const int lane = threadIdx.x & 31;
    const int n    = lane & 15;
    const int mofs = (lane >> 4) * 8;
    __bf16* out = midH + (size_t)h * N2;
    #pragma unroll
    for (int si = 0; si < 2; si++)
        #pragma unroll
        for (int sj = 0; sj < 2; sj++)
            #pragma unroll
            for (int r = 0; r < 8; r++)
                out[(size_t)(i0 + si * 16 + mofs + r) * NN + (j0 + sj * 16 + n)] =
                    (__bf16)acc[si][sj][r];
}

// ---------------------------------------------------------------------------
// Kernel C: gate * mid, output projection (H->C via WMMA), +bias +residual,
// final LayerNorm, f32 output.  16 positions per block, 8 waves.
// midH and gateH share the identical h-major indexing -> fused 16B loads.
// grid = 9216 blocks x 256 threads
// ---------------------------------------------------------------------------
__global__ __launch_bounds__(256)
void kc_out(const float* __restrict__ pair, const __bf16* __restrict__ midH,
            const __bf16* __restrict__ gateH, const __bf16* __restrict__ WoT,
            const float* __restrict__ bias_o, const float* __restrict__ g_out,
            const float* __restrict__ b_out, float* __restrict__ out)
{
    __shared__ __attribute__((aligned(16))) __bf16 am[16][LDA];
    __shared__ float ys[16][LDO];
    const int posbase = blockIdx.x * 16;
    const int t    = threadIdx.x;
    const int wave = t >> 5;
    const int lane = t & 31;

    // Phase 1: gather 128(h) x 16(pos) slab, apply gate, transpose into LDS
    {
        const int h  = t >> 1;
        const int j8 = (t & 1) * 8;
        const size_t off = (size_t)h * N2 + posbase + j8;
        v8bf mv = *(const v8bf*)(midH  + off);
        v8bf gv = *(const v8bf*)(gateH + off);
        #pragma unroll
        for (int r = 0; r < 8; r++)
            am[j8 + r][h] = (__bf16)((float)mv[r] * (float)gv[r]);
    }
    __syncthreads();

    // Phase 2: WMMA projection.  Wave w owns c-tile w, K = H = 128.
    const int cbase = wave * 16;
    v8f acc = {};
    #pragma unroll
    for (int ks = 0; ks < 4; ks++) {
        const int k = ks * 32;
        v16bf fa = load_a_frag(&am[0][0] + k, LDA);
        v16bf fb = load_b_frag(WoT + cbase * HH + k, HH);
        acc = wmma_bf16(fa, fb, acc);
    }

    // bias + residual -> LDS (f32)
    const int n    = lane & 15;
    const int c    = cbase + n;
    const int mofs = (lane >> 4) * 8;
    const float bc = bias_o[c];
    #pragma unroll
    for (int r = 0; r < 8; r++) {
        const int m = mofs + r;
        ys[m][c] = acc[r] + bc + pair[(size_t)(posbase + m) * CC + c];
    }
    __syncthreads();

    // Phase 3: final LayerNorm (2 rows per wave) + coalesced f32 store
    {
        const int row = 2 * wave + (lane >> 4);
        const int c0  = (lane & 15) * 8;
        float s = 0.f, ss = 0.f;
        #pragma unroll
        for (int j = 0; j < 8; j++) { float v = ys[row][c0 + j]; s += v; ss += v * v; }
        #pragma unroll
        for (int m = 1; m < 16; m <<= 1) {
            s  += __shfl_xor(s,  m, 32);
            ss += __shfl_xor(ss, m, 32);
        }
        const float mean = s * (1.f / CC);
        const float rs   = rsqrtf(ss * (1.f / CC) - mean * mean + 1e-5f);
        float tmp[8];
        #pragma unroll
        for (int j = 0; j < 8; j++) {
            const int cc = c0 + j;
            tmp[j] = (ys[row][cc] - mean) * rs * g_out[cc] + b_out[cc];
        }
        float* op = out + (size_t)(posbase + row) * CC + c0;
        *(float4*)op       = make_float4(tmp[0], tmp[1], tmp[2], tmp[3]);
        *(float4*)(op + 4) = make_float4(tmp[4], tmp[5], tmp[6], tmp[7]);
    }
}

// ---------------------------------------------------------------------------
extern "C" void kernel_launch(void* const* d_in, const int* in_sizes, int n_in,
                              void* d_out, int out_size, void* d_ws, size_t ws_size,
                              hipStream_t stream)
{
    const float* pair    = (const float*)d_in[0];
    const float* g_left  = (const float*)d_in[1];
    const float* b_left  = (const float*)d_in[2];
    const float* g_right = (const float*)d_in[3];
    const float* b_right = (const float*)d_in[4];
    const float* g_out   = (const float*)d_in[5];
    const float* b_out   = (const float*)d_in[6];
    const float* W_l     = (const float*)d_in[7];
    const float* bias_l  = (const float*)d_in[8];
    const float* W_r     = (const float*)d_in[9];
    const float* bias_r  = (const float*)d_in[10];
    const float* W_g     = (const float*)d_in[11];
    const float* bias_g  = (const float*)d_in[12];
    const float* W_e     = (const float*)d_in[13];
    const float* bias_e  = (const float*)d_in[14];
    const float* W_o     = (const float*)d_in[15];
    const float* bias_o  = (const float*)d_in[16];
    float* outp = (float*)d_out;

    // Workspace carve-up (all 256B aligned)
    char*  ws = (char*)d_ws;
    size_t o  = 0;
    auto carve = [&](size_t bytes) -> void* {
        void* p = ws + o;
        o += (bytes + 255) & ~(size_t)255;
        return p;
    };
    const size_t wbytes = (size_t)128 * 128 * sizeof(__bf16);
    const size_t tbytes = (size_t)HH * N2 * sizeof(__bf16);   // 37.75 MB each
    __bf16* WlT    = (__bf16*)carve(wbytes);
    __bf16* WrT    = (__bf16*)carve(wbytes);
    __bf16* WgT    = (__bf16*)carve(wbytes);
    __bf16* WeT    = (__bf16*)carve(wbytes);
    __bf16* WoT    = (__bf16*)carve(wbytes);
    __bf16* leftA  = (__bf16*)carve(tbytes);
    __bf16* rightT = (__bf16*)carve(tbytes);
    __bf16* gateH  = (__bf16*)carve(tbytes);
    __bf16* midH   = (__bf16*)carve(tbytes);

    kw_conv<<<5 * 128, 128, 0, stream>>>(W_l, W_r, W_g, W_e, W_o,
                                         WlT, WrT, WgT, WeT, WoT);

    ka_proj<<<N2 / 16, 256, 0, stream>>>(pair, g_left, b_left, g_right, b_right,
                                         bias_l, bias_r, bias_g, bias_e,
                                         WlT, WrT, WgT, WeT,
                                         leftA, rightT, gateH);

    kb_tri<<<(NN / 64) * (NN / 128) * HH, 256, 0, stream>>>(leftA, rightT, midH);

    kc_out<<<N2 / 16, 256, 0, stream>>>(pair, midH, gateH, WoT,
                                        bias_o, g_out, b_out, outp);
}
